// SpatioTemporalGCN_60370060312868
// MI455X (gfx1250) — matrix operand order repaired and checked by reading, not compiled, not run on hardware
//
#include <hip/hip_runtime.h>

// ---------------------------------------------------------------------------
// MI455X (gfx1250) implementation. wave32. All heavy GEMMs (tcn conv1d as
// implicit GEMM, gcn2, output layer) run on v_wmma_f32_16x16x32_bf16.
// Fragments follow the CDNA5 ISA VGPR layouts as contiguous per-lane loads
// (A and B^T: two 16B chunks at K=(l>>4)*8 and +16).
// Conv weights are staged through LDS (4KB tiles) and shared by 8 waves;
// each wave carries 4 f32 accumulators to reuse its A fragment 4x.
// ---------------------------------------------------------------------------

typedef __attribute__((ext_vector_type(8)))  unsigned short ushort8;
typedef __attribute__((ext_vector_type(16))) __bf16        bf16x16;
typedef __attribute__((ext_vector_type(8)))  float         f32x8;

union FragU { ushort8 h[2]; bf16x16 v; };
static_assert(sizeof(FragU) == 32, "frag size");

#define WMMA_BF16(A, B, C) \
  __builtin_amdgcn_wmma_f32_16x16x32_bf16(false, (A), false, (B), (short)0, (C), false, false)

__device__ __forceinline__ unsigned short f32_to_bf16(float f) {
  unsigned int u = __float_as_uint(f);
  u += 0x7FFFu + ((u >> 16) & 1u);   // round-to-nearest-even
  return (unsigned short)(u >> 16);
}
__device__ __forceinline__ float bf16_to_f32(unsigned short h) {
  return __uint_as_float(((unsigned int)h) << 16);
}

// A-fragment: 16x32 (MxK) bf16, row-major source with leading dim ldk.
__device__ __forceinline__ bf16x16 loadA(const unsigned short* base, int ldk, int lane) {
  int m  = lane & 15;
  int k0 = (lane >> 4) * 8;
  const unsigned short* p = base + (size_t)m * ldk + k0;
  FragU u;
  u.h[0] = *(const ushort8*)(p);        // K = k0 .. k0+7
  u.h[1] = *(const ushort8*)(p + 16);   // K = k0+16 .. k0+23
  return u.v;
}
// B-fragment: 32x16 (KxN) bf16 from B^T stored [n][k] row-major (ldk).
__device__ __forceinline__ bf16x16 loadBT(const unsigned short* baseT, int ldk, int lane) {
  int n  = lane & 15;
  int k0 = (lane >> 4) * 8;
  const unsigned short* p = baseT + (size_t)n * ldk + k0;
  FragU u;
  u.h[0] = *(const ushort8*)(p);
  u.h[1] = *(const ushort8*)(p + 16);
  return u.v;
}

// --------------------------- problem constants -----------------------------
constexpr int  B_   = 32;
constexpr int  S_   = 512;
constexpr int  N_   = 6;
constexpr int  CIN  = 12;
constexpr int  H1   = 128;
constexpr int  H_   = 256;
constexpr int  TC   = 1536;
constexpr long ROWS1 = (long)B_ * S_ * N_;   // 98304
constexpr long BS    = (long)B_ * S_;        // 16384
constexpr float EPS  = 1e-5f;

// ----------------------------- small kernels -------------------------------
__global__ void cast_f32_to_bf16_k(const float* __restrict__ in,
                                   unsigned short* __restrict__ out, long n) {
  long i = (long)blockIdx.x * blockDim.x + threadIdx.x;
  if (i < n) out[i] = f32_to_bf16(in[i]);
}

// cw[o][i][t] (f32) -> wt[t][o][i] (bf16)
__global__ void cast_convw_k(const float* __restrict__ cw,
                             unsigned short* __restrict__ wt) {
  long idx = (long)blockIdx.x * blockDim.x + threadIdx.x;
  long total = 5L * TC * TC;
  if (idx >= total) return;
  int  i = (int)(idx % TC);
  long r = idx / TC;
  int  o = (int)(r % TC);
  int  t = (int)(r / TC);
  wt[idx] = f32_to_bf16(cw[((long)o * TC + i) * 5 + t]);
}

// per-channel sum/sumsq over rows (channel-major layout [row][C])
__global__ void stats_k(const float* __restrict__ x, long rows, int C,
                        float* __restrict__ slot) {
  float* sum   = slot;
  float* sumsq = slot + 1536;
  long per = (rows + gridDim.x - 1) / gridDim.x;
  long r0 = (long)blockIdx.x * per;
  long r1 = r0 + per; if (r1 > rows) r1 = rows;
  for (int c = threadIdx.x; c < C; c += blockDim.x) {
    float s = 0.f, s2 = 0.f;
    for (long r = r0; r < r1; ++r) {
      float v = x[r * C + c];
      s += v; s2 += v * v;
    }
    atomicAdd(&sum[c], s);
    atomicAdd(&sumsq[c], s2);
  }
}

__global__ void finalize_stats_k(float* __restrict__ slot, int C, float count) {
  int c = blockIdx.x * blockDim.x + threadIdx.x;
  if (c >= C) return;
  float m = slot[c] / count;
  float v = slot[1536 + c] / count - m * m;
  if (v < 0.f) v = 0.f;
  slot[2 * 1536 + c] = m;               // mean
  slot[3 * 1536 + c] = rsqrtf(v + EPS); // inv-std
}

// ------------------------------- gcn1 (K=12) -------------------------------
__global__ void gcn1_gemm_k(const float* __restrict__ x,
                            const float* __restrict__ w1, const float* __restrict__ b1,
                            const float* __restrict__ rw1, const float* __restrict__ rb1,
                            float* __restrict__ xt, float* __restrict__ res) {
  long idx = (long)blockIdx.x * blockDim.x + threadIdx.x;
  if (idx >= ROWS1 * H1) return;
  long r = idx >> 7;
  int  o = (int)(idx & 127);
  const float* xr = x + r * CIN;
  float a = b1[o], rr = rb1[o];
#pragma unroll
  for (int i = 0; i < CIN; ++i) {
    float v = xr[i];
    a  += v * w1[o * CIN + i];
    rr += v * rw1[o * CIN + i];
  }
  xt[idx] = a;
  res[idx] = rr;
}

// xc[bs][n][o] = sum_m (adj*eimp)[n][m] * xt[bs][m][o]
__global__ void gcn_aggregate_k(const float* __restrict__ adj,
                                const float* __restrict__ eimp,
                                const float* __restrict__ xt,
                                float* __restrict__ xc, int C) {
  long idx = (long)blockIdx.x * blockDim.x + threadIdx.x;
  long total = ROWS1 * C;
  if (idx >= total) return;
  int  o  = (int)(idx % C);
  long r  = idx / C;
  int  n  = (int)(r % N_);
  long bs = r / N_;
  float acc = 0.f;
#pragma unroll
  for (int m = 0; m < N_; ++m)
    acc += adj[n * N_ + m] * eimp[n * N_ + m] * xt[(bs * N_ + m) * C + o];
  xc[idx] = acc;
}

// h = relu(bn(xc)*g+beta + bn(res)*rg+rbeta) -> bf16
__global__ void gcn_bn_relu_bf16_k(const float* __restrict__ xc, const float* __restrict__ res,
                                   const float* __restrict__ slot_xc, const float* __restrict__ slot_res,
                                   const float* __restrict__ g, const float* __restrict__ beta,
                                   const float* __restrict__ rg, const float* __restrict__ rbeta,
                                   unsigned short* __restrict__ out, int C) {
  long idx = (long)blockIdx.x * blockDim.x + threadIdx.x;
  long total = ROWS1 * C;
  if (idx >= total) return;
  int c = (int)(idx % C);
  float a = (xc[idx]  - slot_xc[2 * 1536 + c]) * slot_xc[3 * 1536 + c] * g[c]  + beta[c];
  float b = (res[idx] - slot_res[2 * 1536 + c]) * slot_res[3 * 1536 + c] * rg[c] + rbeta[c];
  float v = a + b;
  out[idx] = f32_to_bf16(v > 0.f ? v : 0.f);
}

// -------------------- gcn2 dual GEMM (WMMA, K=128) -------------------------
// block 128 = 4 waves; wave w covers o in [w*64, w*64+64): 4 subtiles x 2
// weight sets = 8 accumulators, A fragment reused 8x per k-chunk.
__global__ void __launch_bounds__(128)
gcn2_gemm_wmma_k(const unsigned short* __restrict__ h1,   // [ROWS1][128] bf16
                 const unsigned short* __restrict__ w2b,  // [256][128] bf16 (= B^T)
                 const unsigned short* __restrict__ rw2b,
                 const float* __restrict__ b2, const float* __restrict__ rb2,
                 float* __restrict__ xt2, float* __restrict__ res2,
                 float* __restrict__ slot_res) {
  int lane = threadIdx.x & 31;
  int wave = threadIdx.x >> 5;           // 0..3
  long row0 = (long)blockIdx.x * 16;
  int o0 = wave * 64;
  f32x8 accA[4] = {};
  f32x8 accR[4] = {};
#pragma unroll
  for (int kc = 0; kc < H1; kc += 32) {
    bf16x16 a = loadA(h1 + row0 * H1 + kc, H1, lane);
#pragma unroll
    for (int s = 0; s < 4; ++s) {
      bf16x16 bw = loadBT(w2b  + (size_t)(o0 + s * 16) * H1 + kc, H1, lane);
      accA[s] = WMMA_BF16(a, bw, accA[s]);
      bf16x16 br = loadBT(rw2b + (size_t)(o0 + s * 16) * H1 + kc, H1, lane);
      accR[s] = WMMA_BF16(a, br, accR[s]);
    }
  }
  int mh = (lane >> 4) * 8;
#pragma unroll
  for (int s = 0; s < 4; ++s) {
    int o = o0 + s * 16 + (lane & 15);
    float bias = b2[o], rbias = rb2[o];
    float sm = 0.f, s2 = 0.f;
#pragma unroll
    for (int r = 0; r < 8; ++r) {
      long row = row0 + mh + r;
      float va = accA[s][r] + bias;
      float vr = accR[s][r] + rbias;
      xt2[row * H_ + o] = va;
      res2[row * H_ + o] = vr;
      sm += vr; s2 += vr * vr;
    }
    atomicAdd(&slot_res[o], sm);
    atomicAdd(&slot_res[1536 + o], s2);
  }
}

// --------------------- tcn conv1d as implicit WMMA GEMM --------------------
// act: [B][512][1536] bf16 channels-last. wt: [5][1536][1536] = [tap][o][i].
// Block 256 thr = 8 waves: wave w owns l-tile (16 rows) w, all share the
// block's o-range [o0b, o0b+64). Weight tile (64o x 32i bf16 = 4KB) is staged
// once per (tap, k-chunk) into LDS and read back as B fragments via ds loads.
// Each wave holds 4 accumulators, reusing its A fragment 4x. Zero padding at
// the l boundary is a per-lane register select (EXEC stays all-ones for wmma).
__global__ void __launch_bounds__(256)
conv_gemm_wmma_k(const unsigned short* __restrict__ act,
                 const unsigned short* __restrict__ wt,
                 const float* __restrict__ cb,
                 float* __restrict__ y) {
  __shared__ unsigned short wlds[64 * 32];   // 4KB: [o_local][i_local]

  int lane = threadIdx.x & 31;
  int wave = threadIdx.x >> 5;               // 0..7 -> l-tile within block
  int lgrp = blockIdx.x;                     // 0..127 (128 l-rows each)
  int lbase = lgrp * 128;
  int b  = lbase >> 9;                       // 512 rows per batch, groups align
  int l0 = (lbase & 511) + wave * 16;
  int o0b = blockIdx.y * 64;

  // staging role: thread t copies 16B of the weight tile
  int srow = threadIdx.x >> 2;               // 0..63 (o_local)
  int squad = (threadIdx.x & 3) * 8;         // i_local offset

  int m  = lane & 15;
  int k0 = (lane >> 4) * 8;
  const ushort8 z8 = {0, 0, 0, 0, 0, 0, 0, 0};
  f32x8 acc[4] = {};

#pragma unroll
  for (int tap = 0; tap < 5; ++tap) {
    int lrow = l0 + m + tap - 2;
    bool ok = (lrow >= 0) && (lrow < S_);
    const unsigned short* arow = act + ((size_t)b * S_ + (ok ? lrow : 0)) * TC;
    const unsigned short* wrow = wt + ((size_t)tap * TC + o0b + srow) * TC + squad;
    for (int i0 = 0; i0 < TC; i0 += 32) {
      __syncthreads();
      *(ushort8*)&wlds[srow * 32 + squad] = *(const ushort8*)(wrow + i0);
      if (i0 + 32 < TC) __builtin_prefetch(wrow + i0 + 32, 0, 0);
      __syncthreads();

      FragU ua;
      ua.h[0] = *(const ushort8*)(arow + i0 + k0);
      ua.h[1] = *(const ushort8*)(arow + i0 + k0 + 16);
      if (!ok) { ua.h[0] = z8; ua.h[1] = z8; }
#pragma unroll
      for (int s = 0; s < 4; ++s) {
        const unsigned short* p = &wlds[(s * 16 + (lane & 15)) * 32 + (lane >> 4) * 8];
        FragU ub;
        ub.h[0] = *(const ushort8*)(p);
        ub.h[1] = *(const ushort8*)(p + 16);
        acc[s] = WMMA_BF16(ua.v, ub.v, acc[s]);
      }
    }
  }
#pragma unroll
  for (int s = 0; s < 4; ++s) {
    int o = o0b + s * 16 + (lane & 15);
    float bias = cb[o];
#pragma unroll
    for (int r = 0; r < 8; ++r) {
      int l = l0 + (lane >> 4) * 8 + r;
      y[((size_t)b * S_ + l) * TC + o] = acc[s][r] + bias;
    }
  }
}

// relu(bn(y)) -> bf16
__global__ void bn_relu_bf16_k(const float* __restrict__ y, const float* __restrict__ slot,
                               const float* __restrict__ g, const float* __restrict__ beta,
                               unsigned short* __restrict__ out) {
  long idx = (long)blockIdx.x * blockDim.x + threadIdx.x;
  long total = BS * TC;
  if (idx >= total) return;
  int c = (int)(idx % TC);
  float v = (y[idx] - slot[2 * 1536 + c]) * slot[3 * 1536 + c] * g[c] + beta[c];
  out[idx] = f32_to_bf16(v > 0.f ? v : 0.f);
}

// relu(bn(y)) + res(bf16) -> bf16
__global__ void bn_relu_addres_bf16_k(const float* __restrict__ y, const float* __restrict__ slot,
                                      const float* __restrict__ g, const float* __restrict__ beta,
                                      const unsigned short* __restrict__ resb,
                                      unsigned short* __restrict__ out) {
  long idx = (long)blockIdx.x * blockDim.x + threadIdx.x;
  long total = BS * TC;
  if (idx >= total) return;
  int c = (int)(idx % TC);
  float v = (y[idx] - slot[2 * 1536 + c]) * slot[3 * 1536 + c] * g[c] + beta[c];
  v = (v > 0.f ? v : 0.f) + bf16_to_f32(resb[idx]);
  out[idx] = f32_to_bf16(v);
}

// ----------------------- output GEMM (WMMA, K=1536) ------------------------
// block 128 = 4 waves; wave w covers o in [w*64, w*64+64), A reused 4x.
__global__ void __launch_bounds__(128)
out_gemm_wmma_k(const unsigned short* __restrict__ act3,  // [BS][1536] bf16
                const unsigned short* __restrict__ owb,   // [256][1536] bf16 (B^T)
                const float* __restrict__ ob,
                float* __restrict__ out) {
  int lane = threadIdx.x & 31;
  int wave = threadIdx.x >> 5;   // 0..3
  long row0 = (long)blockIdx.x * 16;
  int o0 = wave * 64;
  f32x8 acc[4] = {};
  const unsigned short* abase = act3 + row0 * TC;
  for (int kc = 0; kc < TC; kc += 32) {
    bf16x16 a = loadA(abase + kc, TC, lane);
#pragma unroll
    for (int s = 0; s < 4; ++s) {
      bf16x16 bvec = loadBT(owb + (size_t)(o0 + s * 16) * TC + kc, TC, lane);
      acc[s] = WMMA_BF16(a, bvec, acc[s]);
    }
  }
  int mh = (lane >> 4) * 8;
#pragma unroll
  for (int s = 0; s < 4; ++s) {
    int o = o0 + s * 16 + (lane & 15);
    float bias = ob[o];
#pragma unroll
    for (int r = 0; r < 8; ++r) {
      float v = acc[s][r] + bias;
      out[(row0 + mh + r) * H_ + o] = (v > 0.f ? v : 0.f);
    }
  }
}

// ------------------------------ host driver --------------------------------
static inline long ceil_div(long a, long b) { return (a + b - 1) / b; }

extern "C" void kernel_launch(void* const* d_in, const int* in_sizes, int n_in,
                              void* d_out, int out_size, void* d_ws, size_t ws_size,
                              hipStream_t stream) {
  // inputs (setup_inputs order), all f32
  const float* x     = (const float*)d_in[0];
  const float* adj   = (const float*)d_in[1];
  const float* eimp  = (const float*)d_in[2];
  const float* w1    = (const float*)d_in[3];
  const float* b1    = (const float*)d_in[4];
  const float* rw1   = (const float*)d_in[5];
  const float* rb1   = (const float*)d_in[6];
  const float* rg1   = (const float*)d_in[7];
  const float* rbeta1= (const float*)d_in[8];
  const float* g1    = (const float*)d_in[9];
  const float* beta1 = (const float*)d_in[10];
  const float* w2    = (const float*)d_in[11];
  const float* b2    = (const float*)d_in[12];
  const float* rw2   = (const float*)d_in[13];
  const float* rb2   = (const float*)d_in[14];
  const float* rg2   = (const float*)d_in[15];
  const float* rbeta2= (const float*)d_in[16];
  const float* g2    = (const float*)d_in[17];
  const float* beta2 = (const float*)d_in[18];
  const float* cw1   = (const float*)d_in[19];
  const float* cb1   = (const float*)d_in[20];
  const float* cg1   = (const float*)d_in[21];
  const float* cbeta1= (const float*)d_in[22];
  const float* cw2   = (const float*)d_in[23];
  const float* cb2   = (const float*)d_in[24];
  const float* cg2   = (const float*)d_in[25];
  const float* cbeta2= (const float*)d_in[26];
  const float* ow    = (const float*)d_in[27];
  const float* ob    = (const float*)d_in[28];

  char* ws = (char*)d_ws;
  // workspace bump layout (bytes), lifetimes allow reuse:
  constexpr size_t SZ1   = (size_t)ROWS1 * H1 * 4;   // 50,331,648
  constexpr size_t SZ1H  = (size_t)ROWS1 * H1 * 2;
  constexpr size_t SZ2   = (size_t)ROWS1 * H_ * 4;   // 100,663,296
  constexpr size_t SZHTC = (size_t)BS * TC * 2;      // 50,331,648
  constexpr size_t o_xt1   = 0;
  constexpr size_t o_res1  = o_xt1 + SZ1;
  constexpr size_t o_xc1   = o_res1 + SZ1;
  constexpr size_t o_y     = o_xt1;                  // reuse (xt1+res1 dead)
  constexpr size_t o_h1    = o_xc1 + SZ1;
  constexpr size_t o_xt2   = o_h1 + SZ1H;
  constexpr size_t o_res2  = o_xt2 + SZ2;
  constexpr size_t o_xc2   = o_res2 + SZ2;
  constexpr size_t o_actB  = o_xt2;                  // reuse (xt2 dead)
  constexpr size_t o_act3  = o_res2;                 // reuse (res2 dead)
  constexpr size_t o_htc   = o_xc2 + SZ2;
  constexpr size_t o_w2b   = o_htc + SZHTC;
  constexpr size_t o_rw2b  = o_w2b + 65536;
  constexpr size_t o_wtap1 = o_rw2b + 65536;
  constexpr size_t o_wtap2 = o_wtap1 + (size_t)5 * TC * TC * 2;
  constexpr size_t o_owb   = o_wtap2 + (size_t)5 * TC * TC * 2;
  constexpr size_t o_stats = o_owb + (size_t)H_ * TC * 2;
  constexpr size_t SZ_SLOT = 4 * 1536 * 4;

  float* xt1  = (float*)(ws + o_xt1);
  float* res1 = (float*)(ws + o_res1);
  float* xc1  = (float*)(ws + o_xc1);
  float* y    = (float*)(ws + o_y);
  unsigned short* h1   = (unsigned short*)(ws + o_h1);
  float* xt2  = (float*)(ws + o_xt2);
  float* res2 = (float*)(ws + o_res2);
  float* xc2  = (float*)(ws + o_xc2);
  unsigned short* actB = (unsigned short*)(ws + o_actB);
  unsigned short* act3 = (unsigned short*)(ws + o_act3);
  unsigned short* htc  = (unsigned short*)(ws + o_htc);
  unsigned short* w2b  = (unsigned short*)(ws + o_w2b);
  unsigned short* rw2b = (unsigned short*)(ws + o_rw2b);
  unsigned short* wtap1= (unsigned short*)(ws + o_wtap1);
  unsigned short* wtap2= (unsigned short*)(ws + o_wtap2);
  unsigned short* owb  = (unsigned short*)(ws + o_owb);
  float* slot_res1 = (float*)(ws + o_stats + 0 * SZ_SLOT);
  float* slot_xc1  = (float*)(ws + o_stats + 1 * SZ_SLOT);
  float* slot_res2 = (float*)(ws + o_stats + 2 * SZ_SLOT);
  float* slot_xc2  = (float*)(ws + o_stats + 3 * SZ_SLOT);
  float* slot_bn1  = (float*)(ws + o_stats + 4 * SZ_SLOT);
  float* slot_bn2  = (float*)(ws + o_stats + 5 * SZ_SLOT);

  hipMemsetAsync(ws + o_stats, 0, 6 * SZ_SLOT, stream);

  const int T = 256;
  // weight casts
  cast_f32_to_bf16_k<<<dim3(ceil_div((long)H_ * H1, T)), T, 0, stream>>>(w2, w2b, (long)H_ * H1);
  cast_f32_to_bf16_k<<<dim3(ceil_div((long)H_ * H1, T)), T, 0, stream>>>(rw2, rw2b, (long)H_ * H1);
  cast_f32_to_bf16_k<<<dim3(ceil_div((long)H_ * TC, T)), T, 0, stream>>>(ow, owb, (long)H_ * TC);
  cast_convw_k<<<dim3(ceil_div(5L * TC * TC, T)), T, 0, stream>>>(cw1, wtap1);
  cast_convw_k<<<dim3(ceil_div(5L * TC * TC, T)), T, 0, stream>>>(cw2, wtap2);

  // ---- gcn1 ----
  gcn1_gemm_k<<<dim3(ceil_div(ROWS1 * H1, T)), T, 0, stream>>>(x, w1, b1, rw1, rb1, xt1, res1);
  stats_k<<<dim3(256), T, 0, stream>>>(res1, ROWS1, H1, slot_res1);
  finalize_stats_k<<<dim3(1), H1, 0, stream>>>(slot_res1, H1, (float)ROWS1);
  gcn_aggregate_k<<<dim3(ceil_div(ROWS1 * H1, T)), T, 0, stream>>>(adj, eimp, xt1, xc1, H1);
  stats_k<<<dim3(256), T, 0, stream>>>(xc1, ROWS1, H1, slot_xc1);
  finalize_stats_k<<<dim3(1), H1, 0, stream>>>(slot_xc1, H1, (float)ROWS1);
  gcn_bn_relu_bf16_k<<<dim3(ceil_div(ROWS1 * H1, T)), T, 0, stream>>>(
      xc1, res1, slot_xc1, slot_res1, g1, beta1, rg1, rbeta1, h1, H1);

  // ---- gcn2 (WMMA) ----
  gcn2_gemm_wmma_k<<<dim3((int)(ROWS1 / 16)), 128, 0, stream>>>(
      h1, w2b, rw2b, b2, rb2, xt2, res2, slot_res2);
  finalize_stats_k<<<dim3(1), H_, 0, stream>>>(slot_res2, H_, (float)ROWS1);
  gcn_aggregate_k<<<dim3(ceil_div(ROWS1 * H_, T)), T, 0, stream>>>(adj, eimp, xt2, xc2, H_);
  stats_k<<<dim3(256), T, 0, stream>>>(xc2, ROWS1, H_, slot_xc2);
  finalize_stats_k<<<dim3(1), H_, 0, stream>>>(slot_xc2, H_, (float)ROWS1);
  gcn_bn_relu_bf16_k<<<dim3(ceil_div(ROWS1 * H_, T)), T, 0, stream>>>(
      xc2, res2, slot_xc2, slot_res2, g2, beta2, rg2, rbeta2, htc, H_);
  // htc is [B][S][1536] bf16 channels-last == TCN input & residual

  // ---- tcn conv1 (WMMA implicit GEMM, LDS-staged weights) ----
  conv_gemm_wmma_k<<<dim3(128, 24), 256, 0, stream>>>(htc, wtap1, cb1, y);
  stats_k<<<dim3(256), T, 0, stream>>>(y, BS, TC, slot_bn1);
  finalize_stats_k<<<dim3(6), T, 0, stream>>>(slot_bn1, TC, (float)BS);
  bn_relu_bf16_k<<<dim3(ceil_div(BS * TC, T)), T, 0, stream>>>(y, slot_bn1, cg1, cbeta1, actB);

  // ---- tcn conv2 ----
  conv_gemm_wmma_k<<<dim3(128, 24), 256, 0, stream>>>(actB, wtap2, cb2, y);
  stats_k<<<dim3(256), T, 0, stream>>>(y, BS, TC, slot_bn2);
  finalize_stats_k<<<dim3(6), T, 0, stream>>>(slot_bn2, TC, (float)BS);
  bn_relu_addres_bf16_k<<<dim3(ceil_div(BS * TC, T)), T, 0, stream>>>(
      y, slot_bn2, cg2, cbeta2, htc, act3);

  // ---- output layer (WMMA) + ReLU ----
  out_gemm_wmma_k<<<dim3((int)(BS / 16)), 128, 0, stream>>>(act3, owb, ob, (float*)d_out);
}
